// Network_85263690760945
// MI455X (gfx1250) — compile-verified, matrix operand order checked by reading, not run on hardware
//
#include <hip/hip_runtime.h>

typedef _Float16 v16h __attribute__((ext_vector_type(16)));
typedef _Float16 v8h  __attribute__((ext_vector_type(8)));
typedef float    v8f  __attribute__((ext_vector_type(8)));

constexpr int B_SZ = 4096;   // batch (GEMM M)
constexpr int D_IN = 2048;   // input dim (GEMM1 K)
constexpr int H    = 2048;   // hidden (GEMM N / GEMM2 K)
constexpr int A_SZ = 5;      // action dim
constexpr int T_ST = 10;     // LIF timesteps

// ---------------------------------------------------------------------------
// fp32 -> fp16 convert (x)
// ---------------------------------------------------------------------------
__global__ void convert_f16(const float* __restrict__ in,
                            _Float16* __restrict__ out, int n) {
  int i = blockIdx.x * blockDim.x + threadIdx.x;
  int stride = gridDim.x * blockDim.x;
  for (; i < n; i += stride) out[i] = (_Float16)in[i];
}

// ---------------------------------------------------------------------------
// W [K][N] fp32 row-major -> Wt [N][K] fp16 ("column-major B" for WMMA)
// ---------------------------------------------------------------------------
__global__ void transpose_f16(const float* __restrict__ W,
                              _Float16* __restrict__ Wt, int K, int N) {
  __shared__ _Float16 tile[32][33];
  const int nb = blockIdx.x * 32, kb = blockIdx.y * 32;
  for (int i = threadIdx.y; i < 32; i += 8)
    tile[i][threadIdx.x] = (_Float16)W[(size_t)(kb + i) * N + nb + threadIdx.x];
  __syncthreads();
  for (int i = threadIdx.y; i < 32; i += 8)
    Wt[(size_t)(nb + i) * K + kb + threadIdx.x] = tile[threadIdx.x][i];
}

// ---------------------------------------------------------------------------
// Async global->LDS copy helpers (CDNA5 GLOBAL_LOAD_ASYNC_TO_LDS_B128, GVS
// addressing: SGPR64 base + VGPR32 unsigned byte offset + imm offset).
// ldsAddr = LDS byte address (low 32 bits of generic shared pointer).
// ---------------------------------------------------------------------------
__device__ __forceinline__ void async_b128_0(uint32_t ldsAddr, uint32_t gOff,
                                             const void* base) {
  asm volatile("global_load_async_to_lds_b128 %0, %1, %2"
               :
               : "v"(ldsAddr), "v"(gOff), "s"(base)
               : "memory");
}
__device__ __forceinline__ void async_b128_16(uint32_t ldsAddr, uint32_t gOff,
                                              const void* base) {
  asm volatile("global_load_async_to_lds_b128 %0, %1, %2 offset:16"
               :
               : "v"(ldsAddr), "v"(gOff), "s"(base)
               : "memory");
}
__device__ __forceinline__ void wait_async0() {
  asm volatile("s_wait_asynccnt 0x0" ::: "memory");
}

// ---------------------------------------------------------------------------
// WMMA GEMM: C = A @ B + bias.  A: [M][K] f16 row-major, Bt: [N][K] f16.
// Block tile 128x128, K-step 32, double-buffered LDS filled by async copies.
// 8 waves: 2 (M) x 4 (N); each wave owns a 64x32 sub-block = 4x2 accumulators
// of v_wmma_f32_16x16x32_f16.
// ---------------------------------------------------------------------------
constexpr int M_BLK = 128, N_BLK = 128, K_BLK = 32;
constexpr int LDSS  = 40;  // 32 + 8 half pad (80B row stride, 16B aligned)

template <bool OUT_F16>
__global__ __launch_bounds__(256) void gemm_wmma(
    const _Float16* __restrict__ A, const _Float16* __restrict__ Bt,
    const float* __restrict__ bias, _Float16* __restrict__ outH,
    float* __restrict__ outF, int M, int N, int K) {
  __shared__ _Float16 As[2][M_BLK * LDSS];
  __shared__ _Float16 Bs[2][N_BLK * LDSS];

  const int tid  = threadIdx.x;
  const int lane = tid & 31;
  const int wave = tid >> 5;     // 0..7
  const int wm   = wave & 1;     // wave row (2 along M)
  const int wn   = wave >> 1;    // wave col (4 along N)

  const int tileM = blockIdx.y * M_BLK;
  const int tileN = blockIdx.x * N_BLK;

  // async staging: each thread owns 16 contiguous halves (32B) of A and of B
  const int ldRow = tid >> 1;          // 0..127
  const int ldCol = (tid & 1) * 16;    // 0 or 16 halves

  const int ldsIdx = ldRow * LDSS + ldCol;
  uint32_t aLds[2], bLds[2];
  aLds[0] = (uint32_t)(uintptr_t)&As[0][ldsIdx];
  aLds[1] = (uint32_t)(uintptr_t)&As[1][ldsIdx];
  bLds[0] = (uint32_t)(uintptr_t)&Bs[0][ldsIdx];
  bLds[1] = (uint32_t)(uintptr_t)&Bs[1][ldsIdx];

  uint32_t aOff = ((uint32_t)(tileM + ldRow) * (uint32_t)K + ldCol) * 2u;
  uint32_t bOff = ((uint32_t)(tileN + ldRow) * (uint32_t)K + ldCol) * 2u;

  const int laneRow = lane & 15;       // M index (A) / N index (B,C/D)
  const int laneHi  = lane >> 4;       // half-wave select
  const int fragOff = laneHi * 8;      // K-half offset per ISA 16-bit A layout

  v8f acc[4][2] = {};

  auto issue = [&](int buf) {
    async_b128_0(aLds[buf], aOff, A);
    async_b128_16(aLds[buf], aOff, A);
    async_b128_0(bLds[buf], bOff, Bt);
    async_b128_16(bLds[buf], bOff, Bt);
    aOff += 2u * K_BLK;
    bOff += 2u * K_BLK;
  };

  // prologue: fill buffer 0
  issue(0);
  wait_async0();
  __syncthreads();

  const int ksteps = K / K_BLK;
  for (int ks = 0; ks < ksteps; ++ks) {
    const int cur = ks & 1;
    if (ks + 1 < ksteps) issue(cur ^ 1);  // overlap next tile with compute

    v16h afrag[4], bfrag[2];
#pragma unroll
    for (int m = 0; m < 4; ++m) {
      const _Float16* p = &As[cur][(wm * 64 + m * 16 + laneRow) * LDSS + fragOff];
      v8h lo = *(const v8h*)(p);        // K = fragOff .. +7
      v8h hi = *(const v8h*)(p + 16);   // K = 16+fragOff .. +7
      afrag[m] = __builtin_shufflevector(lo, hi, 0, 1, 2, 3, 4, 5, 6, 7, 8, 9,
                                         10, 11, 12, 13, 14, 15);
    }
#pragma unroll
    for (int n = 0; n < 2; ++n) {
      const _Float16* p = &Bs[cur][(wn * 32 + n * 16 + laneRow) * LDSS + fragOff];
      v8h lo = *(const v8h*)(p);
      v8h hi = *(const v8h*)(p + 16);
      bfrag[n] = __builtin_shufflevector(lo, hi, 0, 1, 2, 3, 4, 5, 6, 7, 8, 9,
                                         10, 11, 12, 13, 14, 15);
    }

#pragma unroll
    for (int m = 0; m < 4; ++m)
#pragma unroll
      for (int n = 0; n < 2; ++n)
        acc[m][n] = __builtin_amdgcn_wmma_f32_16x16x32_f16(
            false, afrag[m], false, bfrag[n], (short)0, acc[m][n], false,
            false);

    // next tile must be resident in LDS before anyone reads it
    wait_async0();
    __syncthreads();
  }

  // epilogue: C/D layout -> row = base + 8*laneHi + r, col = base + laneRow
#pragma unroll
  for (int n = 0; n < 2; ++n) {
    const int col = tileN + wn * 32 + n * 16 + laneRow;
    const float bv = bias[col];
#pragma unroll
    for (int m = 0; m < 4; ++m) {
#pragma unroll
      for (int r = 0; r < 8; ++r) {
        const int row = tileM + wm * 64 + m * 16 + laneHi * 8 + r;
        const float v = acc[m][n][r] + bv;
        if (OUT_F16)
          outH[(size_t)row * N + col] = (_Float16)v;
        else
          outF[(size_t)row * N + col] = v;
      }
    }
  }
}

// ---------------------------------------------------------------------------
// Fused LIF scan + firing-rate, elementwise in place over cur.
// v <- (v+cur)/2 ; spike if v>=1 then hard reset.
// ---------------------------------------------------------------------------
__global__ void lif_rate(float* __restrict__ cur, int n) {
  int i = blockIdx.x * blockDim.x + threadIdx.x;
  int stride = gridDim.x * blockDim.x;
  for (; i < n; i += stride) {
    const float c = cur[i];
    float v = 0.f, sum = 0.f;
#pragma unroll
    for (int t = 0; t < T_ST; ++t) {
      v = 0.5f * (v + c);
      if (v >= 1.0f) {
        sum += 1.0f;
        v = 0.0f;
      }
    }
    cur[i] = sum * (1.0f / T_ST);
  }
}

// ---------------------------------------------------------------------------
// Head: pre = rate @ W_out + b_out; out-LIF with v0=0 => s = (pre >= 2).
// One wave per batch row, wave32 shuffle reduction.
// ---------------------------------------------------------------------------
__global__ __launch_bounds__(256) void head_kernel(
    const float* __restrict__ rate, const float* __restrict__ Wout,
    const float* __restrict__ bout, float* __restrict__ out) {
  const int wave = threadIdx.x >> 5, lane = threadIdx.x & 31;
  const int row = blockIdx.x * 8 + wave;
  const float* r = rate + (size_t)row * H;
  float acc[A_SZ] = {};
  for (int k = lane; k < H; k += 32) {
    const float rv = r[k];
#pragma unroll
    for (int a = 0; a < A_SZ; ++a) acc[a] += rv * Wout[k * A_SZ + a];
  }
#pragma unroll
  for (int a = 0; a < A_SZ; ++a)
    for (int off = 16; off; off >>= 1) acc[a] += __shfl_down(acc[a], off, 32);
  if (lane == 0) {
#pragma unroll
    for (int a = 0; a < A_SZ; ++a) {
      const float pre = acc[a] + bout[a];
      out[(size_t)row * A_SZ + a] = (pre * 0.5f - 1.0f >= 0.0f) ? 1.0f : 0.0f;
    }
  }
}

// ---------------------------------------------------------------------------
extern "C" void kernel_launch(void* const* d_in, const int* in_sizes, int n_in,
                              void* d_out, int out_size, void* d_ws,
                              size_t ws_size, hipStream_t stream) {
  const float* x     = (const float*)d_in[0];
  const float* W_in  = (const float*)d_in[1];
  const float* b_in  = (const float*)d_in[2];
  const float* W_snn = (const float*)d_in[3];
  const float* b_snn = (const float*)d_in[4];
  const float* W_out = (const float*)d_in[5];
  const float* b_out = (const float*)d_in[6];
  float* out = (float*)d_out;

  char* ws = (char*)d_ws;
  _Float16* xh    = (_Float16*)(ws);                        // 16 MB
  _Float16* WtIn  = (_Float16*)(ws + ((size_t)16 << 20));   //  8 MB
  _Float16* WtSnn = (_Float16*)(ws + ((size_t)24 << 20));   //  8 MB
  _Float16* hF16  = (_Float16*)(ws + ((size_t)32 << 20));   // 16 MB
  float*    cur   = (float*)   (ws + ((size_t)48 << 20));   // 32 MB (80 MB total)

  convert_f16<<<2048, 256, 0, stream>>>(x, xh, B_SZ * D_IN);
  transpose_f16<<<dim3(H / 32, D_IN / 32), dim3(32, 8), 0, stream>>>(
      W_in, WtIn, D_IN, H);
  transpose_f16<<<dim3(H / 32, H / 32), dim3(32, 8), 0, stream>>>(
      W_snn, WtSnn, H, H);

  dim3 grid(H / N_BLK, B_SZ / M_BLK);  // 16 x 32 blocks
  gemm_wmma<true><<<grid, 256, 0, stream>>>(xh, WtIn, b_in, hF16, nullptr,
                                            B_SZ, H, D_IN);
  gemm_wmma<false><<<grid, 256, 0, stream>>>(hF16, WtSnn, b_snn, nullptr, cur,
                                             B_SZ, H, H);

  lif_rate<<<2048, 256, 0, stream>>>(cur, B_SZ * H);
  head_kernel<<<B_SZ / 8, 256, 0, stream>>>(cur, W_out, b_out, out);
}
